// RWKVTimeMixing_20572893348158
// MI455X (gfx1250) — matrix operand order repaired and checked by reading, not compiled, not run on hardware
//
#include <hip/hip_runtime.h>

typedef __attribute__((ext_vector_type(16))) __bf16 v16bf;
typedef __attribute__((ext_vector_type(8)))  __bf16 v8bf;
typedef __attribute__((ext_vector_type(4)))  __bf16 v4bf;
typedef __attribute__((ext_vector_type(8)))  float  v8f;
typedef __attribute__((ext_vector_type(4)))  float  v4f;

#define Bsz 4
#define Tsz 4096
#define Csz 1024
#define Msz (Bsz * Tsz)          // 16384 rows for all GEMMs

// ---------------------------------------------------------------------------
// f32 -> bf16 conversion (vectorized x4)
// ---------------------------------------------------------------------------
__global__ __launch_bounds__(256) void cvt_bf16_kernel(const float* __restrict__ src,
                                                       __bf16* __restrict__ dst, int n4) {
  int i = blockIdx.x * blockDim.x + threadIdx.x;
  if (i < n4) {
    v4f f = ((const v4f*)src)[i];
    ((v4bf*)dst)[i] = __builtin_convertvector(f, v4bf);
  }
}

// ---------------------------------------------------------------------------
// Fused K/V/R projection GEMM: [16384 x 1024] x 3x[1024 x 1024]^T
// One wave computes a 16(M) x 32(N) strip for all three weights.
// A fragment shared by 6 WMMAs per k-step. Sigmoid fused into r store.
// ---------------------------------------------------------------------------
__global__ __launch_bounds__(256) void kvr_gemm_kernel(
    const __bf16* __restrict__ xb,
    const __bf16* __restrict__ kw, const __bf16* __restrict__ vw,
    const __bf16* __restrict__ rw,
    float* __restrict__ kf, float* __restrict__ vf, float* __restrict__ rf) {
  const int wid  = blockIdx.x * 8 + (threadIdx.x >> 5);
  const int lane = threadIdx.x & 31;
  const int l  = lane & 15;
  const int hi = lane >> 4;
  const int mt = wid >> 5;        // 0..1023  (M tile)
  const int ng = wid & 31;        // 0..31    (N group of 2 tiles)
  const int m0 = mt << 4;
  const int n0 = ng << 5;

  v8f acc[3][2] = {};
  const __bf16* aRow = xb + (size_t)(m0 + l) * Csz;

  for (int k0 = 0; k0 < Csz; k0 += 32) {
    // A fragment: lane(l,hi) -> row m0+l, K {k0+16hi..+15}
    v16bf A = *(const v16bf*)(aRow + k0 + hi * 16);
    const int bc = k0 + hi * 16;
#pragma unroll
    for (int nt = 0; nt < 2; ++nt) {
      const size_t brow = (size_t)(n0 + nt * 16 + l) * Csz + bc;
      v16bf Bk = *(const v16bf*)(kw + brow);
      v16bf Bv = *(const v16bf*)(vw + brow);
      v16bf Br = *(const v16bf*)(rw + brow);
      acc[0][nt] = __builtin_amdgcn_wmma_f32_16x16x32_bf16(false, A, false, Bk, (short)0, acc[0][nt], false, false);
      acc[1][nt] = __builtin_amdgcn_wmma_f32_16x16x32_bf16(false, A, false, Bv, (short)0, acc[1][nt], false, false);
      acc[2][nt] = __builtin_amdgcn_wmma_f32_16x16x32_bf16(false, A, false, Br, (short)0, acc[2][nt], false, false);
    }
  }
#pragma unroll
  for (int nt = 0; nt < 2; ++nt) {
    const int dcol = n0 + nt * 16 + l;
#pragma unroll
    for (int i = 0; i < 8; ++i) {
      const size_t o = (size_t)(m0 + hi * 8 + i) * Csz + dcol;
      kf[o] = acc[0][nt][i];
      vf[o] = acc[1][nt][i];
      rf[o] = 1.0f / (1.0f + __expf(-acc[2][nt][i]));  // fused sigmoid
    }
  }
}

// ---------------------------------------------------------------------------
// Sequential WKV scan: one thread per (b,c) channel, T=4096 steps.
// 128 single-wave blocks spread across WGPs (latency-bound stage).
// Fuses r * wkv and bf16 downcast for the output GEMM; writes final state.
// ---------------------------------------------------------------------------
__global__ __launch_bounds__(32) void wkv_scan_kernel(
    const float* __restrict__ kf, const float* __restrict__ vf,
    const float* __restrict__ rf,
    const float* __restrict__ time_decay, const float* __restrict__ time_first,
    const float* __restrict__ aa0, const float* __restrict__ bb0,
    const float* __restrict__ pp0,
    __bf16* __restrict__ rwkvb, float* __restrict__ state_out) {
  const int tid = blockIdx.x * 32 + threadIdx.x;   // 0..B*C-1
  const int b = tid >> 10;
  const int c = tid & (Csz - 1);

  const float w = __expf(time_decay[c]);
  const float u = time_first[c];
  float aa = aa0[tid], bb = bb0[tid], pp = pp0[tid];

  size_t idx = (size_t)b * Tsz * Csz + c;
  for (int t = 0; t < Tsz; ++t, idx += Csz) {
    __builtin_prefetch(kf + idx + 16 * Csz, 0, 1);   // global_prefetch_b8
    __builtin_prefetch(vf + idx + 16 * Csz, 0, 1);
    const float kt = kf[idx];
    const float vt = vf[idx];
    const float rt = rf[idx];
    // output weighting (log-sum-exp stabilized)
    const float ww = u + kt;
    const float p  = fmaxf(pp, ww);
    const float e1 = __expf(pp - p);
    const float e2 = __expf(ww - p);
    const float wkv = (e1 * aa + e2 * vt) / (e1 * bb + e2);
    rwkvb[idx] = (__bf16)(rt * wkv);
    // state update
    const float ww2 = pp - w;
    const float p2  = fmaxf(ww2, kt);
    const float e1b = __expf(ww2 - p2);
    const float e2b = __expf(kt - p2);
    aa = e1b * aa + e2b * vt;
    bb = e1b * bb + e2b;
    pp = p2;
  }
  state_out[tid]                 = aa;
  state_out[Bsz * Csz + tid]     = bb;
  state_out[2 * Bsz * Csz + tid] = pp;
}

// ---------------------------------------------------------------------------
// Output projection GEMM: out = (r*wkv) x output_w^T, f32 result to d_out.
// ---------------------------------------------------------------------------
__global__ __launch_bounds__(256) void out_gemm_kernel(
    const __bf16* __restrict__ ab, const __bf16* __restrict__ ow,
    float* __restrict__ out) {
  const int wid  = blockIdx.x * 8 + (threadIdx.x >> 5);
  const int lane = threadIdx.x & 31;
  const int l  = lane & 15;
  const int hi = lane >> 4;
  const int mt = wid >> 5;
  const int ng = wid & 31;
  const int m0 = mt << 4;
  const int n0 = ng << 5;

  v8f acc[2] = {};
  const __bf16* aRow = ab + (size_t)(m0 + l) * Csz;

  for (int k0 = 0; k0 < Csz; k0 += 32) {
    v16bf A = *(const v16bf*)(aRow + k0 + hi * 16);
    const int bc = k0 + hi * 16;
#pragma unroll
    for (int nt = 0; nt < 2; ++nt) {
      const size_t brow = (size_t)(n0 + nt * 16 + l) * Csz + bc;
      v16bf Bw = *(const v16bf*)(ow + brow);
      acc[nt] = __builtin_amdgcn_wmma_f32_16x16x32_bf16(false, A, false, Bw, (short)0, acc[nt], false, false);
    }
  }
#pragma unroll
  for (int nt = 0; nt < 2; ++nt) {
    const int dcol = n0 + nt * 16 + l;
#pragma unroll
    for (int i = 0; i < 8; ++i) {
      out[(size_t)(m0 + hi * 8 + i) * Csz + dcol] = acc[nt][i];
    }
  }
}

// ---------------------------------------------------------------------------
extern "C" void kernel_launch(void* const* d_in, const int* in_sizes, int n_in,
                              void* d_out, int out_size, void* d_ws, size_t ws_size,
                              hipStream_t stream) {
  const float* x   = (const float*)d_in[0];
  const float* kW  = (const float*)d_in[1];
  const float* vW  = (const float*)d_in[2];
  const float* rW  = (const float*)d_in[3];
  const float* oW  = (const float*)d_in[4];
  const float* td  = (const float*)d_in[5];
  const float* tf  = (const float*)d_in[6];
  const float* aa0 = (const float*)d_in[7];
  const float* bb0 = (const float*)d_in[8];
  const float* pp0 = (const float*)d_in[9];
  float* out = (float*)d_out;

  // Workspace layout (all offsets 256B aligned)
  char* ws = (char*)d_ws;
  const size_t XBF   = (size_t)Msz * Csz * 2;       // 32 MB  x in bf16
  const size_t WBF   = (size_t)Csz * Csz * 2;       //  2 MB  each weight bf16
  const size_t KVF   = (size_t)Msz * Csz * 4;       // 64 MB  each f32 intermediate
  __bf16* xb    = (__bf16*)(ws);
  __bf16* kwb   = (__bf16*)(ws + XBF);
  __bf16* vwb   = (__bf16*)(ws + XBF + WBF);
  __bf16* rwb   = (__bf16*)(ws + XBF + 2 * WBF);
  __bf16* owb   = (__bf16*)(ws + XBF + 3 * WBF);
  float*  kf    = (float*)(ws + XBF + 4 * WBF);
  float*  vf    = (float*)(ws + XBF + 4 * WBF + KVF);
  float*  rf    = (float*)(ws + XBF + 4 * WBF + 2 * KVF);
  __bf16* rwkvb = (__bf16*)(ws + XBF + 4 * WBF + 3 * KVF);

  // 1. Convert x and weights to bf16
  {
    int n4 = (Msz * Csz) / 4;                       // 4,194,304
    cvt_bf16_kernel<<<n4 / 256, 256, 0, stream>>>(x, xb, n4);
    int w4 = (Csz * Csz) / 4;                       // 262,144
    cvt_bf16_kernel<<<w4 / 256, 256, 0, stream>>>(kW, kwb, w4);
    cvt_bf16_kernel<<<w4 / 256, 256, 0, stream>>>(vW, vwb, w4);
    cvt_bf16_kernel<<<w4 / 256, 256, 0, stream>>>(rW, rwb, w4);
    cvt_bf16_kernel<<<w4 / 256, 256, 0, stream>>>(oW, owb, w4);
  }

  // 2. Fused k/v/r projections via WMMA (32768 waves = 4096 blocks x 8 waves)
  kvr_gemm_kernel<<<4096, 256, 0, stream>>>(xb, kwb, vwb, rwb, kf, vf, rf);

  // 3. Sequential WKV scan (4096 channels, final states to tail of d_out)
  wkv_scan_kernel<<<Bsz * Csz / 32, 32, 0, stream>>>(
      kf, vf, rf, td, tf, aa0, bb0, pp0, rwkvb, out + (size_t)Msz * Csz);

  // 4. Output projection via WMMA
  out_gemm_kernel<<<4096, 256, 0, stream>>>(rwkvb, owb, out);
}
